// Attention_57028575756535
// MI455X (gfx1250) — matrix operand order repaired and checked by reading, not compiled, not run on hardware
//
#include <hip/hip_runtime.h>
#include <hip/hip_bf16.h>

typedef __bf16 bf16_t;
typedef __attribute__((ext_vector_type(8)))  __bf16 v8bf;
typedef __attribute__((ext_vector_type(16))) __bf16 v16bf;
typedef __attribute__((ext_vector_type(8)))  float  v8f;
typedef __attribute__((__vector_size__(4 * sizeof(int)))) int v4i;

#define BATCH   2
#define S_LEN   2048
#define HDIM    2048
#define NHEADS  16
#define HEADD   128

__device__ __forceinline__ v16bf bf_cat(v8bf lo, v8bf hi) {
    return __builtin_shufflevector(lo, hi, 0,1,2,3,4,5,6,7,8,9,10,11,12,13,14,15);
}

__device__ __forceinline__ v8f wmma_bf16(v16bf a, v16bf b, v8f c) {
    // D = A(16x32 bf16) * B(32x16 bf16) + C(16x16 f32)
    return __builtin_amdgcn_wmma_f32_16x16x32_bf16(false, a, false, b, (short)0, c, false, false);
}

// ---------------------------------------------------------------------------
// CDNA5 async global->LDS copy (ASYNCcnt path)
// ---------------------------------------------------------------------------
__device__ __forceinline__ void async_ld_b128(void* lds, const void* g) {
#if __has_builtin(__builtin_amdgcn_global_load_async_to_lds_b128)
    __builtin_amdgcn_global_load_async_to_lds_b128(
        (__attribute__((address_space(1))) v4i*)(g),
        (__attribute__((address_space(3))) v4i*)(lds), 0, 0);
#else
    unsigned int loff = (unsigned int)(unsigned long long)(lds);
    asm volatile("global_load_async_to_lds_b128 %0, %1, off"
                 :: "v"(loff), "v"(g) : "memory");
#endif
}

__device__ __forceinline__ void wait_async0() {
#if __has_builtin(__builtin_amdgcn_s_wait_asynccnt)
    __builtin_amdgcn_s_wait_asynccnt(0);
#else
    asm volatile("s_wait_asynccnt 0x0" ::: "memory");
#endif
}
__device__ __forceinline__ void wait_async3() {
#if __has_builtin(__builtin_amdgcn_s_wait_asynccnt)
    __builtin_amdgcn_s_wait_asynccnt(3);
#else
    asm volatile("s_wait_asynccnt 0x3" ::: "memory");
#endif
}
__device__ __forceinline__ void wait_async4() {
#if __has_builtin(__builtin_amdgcn_s_wait_asynccnt)
    __builtin_amdgcn_s_wait_asynccnt(4);
#else
    asm volatile("s_wait_asynccnt 0x4" ::: "memory");
#endif
}

// ---------------------------------------------------------------------------
// fp32 -> bf16 conversion
// ---------------------------------------------------------------------------
__global__ void __launch_bounds__(256) f32_to_bf16_kernel(const float* __restrict__ src,
                                                          bf16_t* __restrict__ dst, int n) {
    int i = (blockIdx.x * blockDim.x + threadIdx.x) * 4;
    if (i + 3 < n) {
        float4 v = *(const float4*)&src[i];
        dst[i + 0] = (bf16_t)v.x;
        dst[i + 1] = (bf16_t)v.y;
        dst[i + 2] = (bf16_t)v.z;
        dst[i + 3] = (bf16_t)v.w;
    }
}

// ---------------------------------------------------------------------------
// 128x64-tile WMMA GEMM:  C[M,N] = A[M,K] * W[N,K]^T  (A, W bf16 row-major)
// Double-buffered LDS staging via async global->LDS loads; peeled tail so the
// steady-state loop has an unconditional issue + s_wait_asynccnt<=3.
// MODE 0: out bf16 at [b, h, s, d]   (Q / K head layout)
// MODE 1: out f32  at [m, n]         (final projection)
// MODE 2: out bf16 at [b, h, d, s]   (V transposed for P@V)
// ---------------------------------------------------------------------------
template <int MODE>
__global__ void __launch_bounds__(256) gemm_wmma_kernel(const bf16_t* __restrict__ A,
                                                        const bf16_t* __restrict__ W,
                                                        void* __restrict__ out,
                                                        int M, int N, int K) {
    __shared__ __align__(16) bf16_t Abuf[2][128 * 32];
    __shared__ __align__(16) bf16_t Bbuf[2][64 * 32];

    const int tid  = threadIdx.x;
    const int lane = tid & 31;
    const int wave = tid >> 5;
    const int n0   = blockIdx.x * 64;
    const int m0   = blockIdx.y * 128;
    const int mtb  = (wave & 3) * 2;   // two 16-row tiles
    const int ntb  = (wave >> 2) * 2;  // two 16-col tiles

    v8f acc[2][2] = {{{}, {}}, {{}, {}}};

    // staging assignment: A: 128x32 (2 x b128/thread), B: 64x32 (1 x b128/thread)
    const int ar = tid >> 1;
    const int ac = (tid & 1) * 16;
    const int br = tid >> 2;
    const int bc = (tid & 3) * 8;
    const bf16_t* Arow = &A[(size_t)(m0 + ar) * K + ac];
    const bf16_t* Wrow = &W[(size_t)(n0 + br) * K + bc];

    auto issue_tile = [&](int kb, int buf) {
        async_ld_b128(&Abuf[buf][ar * 32 + ac],     Arow + kb);
        async_ld_b128(&Abuf[buf][ar * 32 + ac + 8], Arow + kb + 8);
        async_ld_b128(&Bbuf[buf][br * 32 + bc],     Wrow + kb);
    };

    auto compute_tile = [&](int cur) {
        // A frag: lane l -> row m=l%16, 16 contiguous K from (l/16)*16
        int ak  = (lane >> 4) * 16;
        int ar0 = (mtb + 0) * 16 + (lane & 15);
        int ar1 = (mtb + 1) * 16 + (lane & 15);
        v16bf av[2];
        av[0] = bf_cat(*(v8bf*)&Abuf[cur][ar0 * 32 + ak], *(v8bf*)&Abuf[cur][ar0 * 32 + ak + 8]);
        av[1] = bf_cat(*(v8bf*)&Abuf[cur][ar1 * 32 + ak], *(v8bf*)&Abuf[cur][ar1 * 32 + ak + 8]);
        // B frag: lane l -> col n=l%16, 16 contiguous K from (l/16)*16
        int bk  = (lane >> 4) * 16;
        int bn0 = (ntb + 0) * 16 + (lane & 15);
        int bn1 = (ntb + 1) * 16 + (lane & 15);
        v16bf bv[2];
        bv[0] = bf_cat(*(v8bf*)&Bbuf[cur][bn0 * 32 + bk], *(v8bf*)&Bbuf[cur][bn0 * 32 + bk + 8]);
        bv[1] = bf_cat(*(v8bf*)&Bbuf[cur][bn1 * 32 + bk], *(v8bf*)&Bbuf[cur][bn1 * 32 + bk + 8]);
#pragma unroll
        for (int i = 0; i < 2; ++i)
#pragma unroll
            for (int j = 0; j < 2; ++j)
                acc[i][j] = wmma_bf16(av[i], bv[j], acc[i][j]);
    };

    // prologue: stage tile 0 into buffer 0 (3 async ops / thread / tile)
    issue_tile(0, 0);

    int kb = 0;
    for (; kb < K - 32; kb += 32) {
        const int cur = (kb >> 5) & 1;
        issue_tile(kb + 32, cur ^ 1);   // stage next tile into the other buffer
        wait_async3();                  // wait only for tile `cur`
        __syncthreads();
        compute_tile(cur);
        __syncthreads();
    }
    // peeled tail: last tile
    {
        const int cur = (kb >> 5) & 1;
        wait_async0();
        __syncthreads();
        compute_tile(cur);
    }

    // epilogue: D layout — lane group (l/16) holds rows r / 8+r, col = l%16
#pragma unroll
    for (int i = 0; i < 2; ++i) {
        int rbase = (mtb + i) * 16 + (lane >> 4) * 8;
#pragma unroll
        for (int j = 0; j < 2; ++j) {
            int ncol = n0 + (ntb + j) * 16 + (lane & 15);
#pragma unroll
            for (int r = 0; r < 8; ++r) {
                int m = m0 + rbase + r;
                float v = acc[i][j][r];
                if (MODE == 1) {
                    ((float*)out)[(size_t)m * N + ncol] = v;
                } else {
                    int b = m / S_LEN, s = m % S_LEN;
                    int h = ncol / HEADD, d = ncol % HEADD;
                    if (MODE == 0)
                        ((bf16_t*)out)[((size_t)(b * NHEADS + h) * S_LEN + s) * HEADD + d] = (bf16_t)v;
                    else // MODE 2: V transposed [b,h,d,s]
                        ((bf16_t*)out)[((size_t)(b * NHEADS + h) * HEADD + d) * S_LEN + s] = (bf16_t)v;
                }
            }
        }
    }
}

// ---------------------------------------------------------------------------
// RoPE applied in place on [b, h, s, d] bf16 tensor (Q or K)
// ---------------------------------------------------------------------------
__global__ void __launch_bounds__(256) rope_kernel(bf16_t* __restrict__ t,
                                                   const float* __restrict__ cosb,
                                                   const float* __restrict__ sinb,
                                                   int total /* B*NH*S*64 */) {
    int i = blockIdx.x * blockDim.x + threadIdx.x;
    if (i >= total) return;
    int d = i & 63;
    int row = i >> 6;                 // (b*NH + h)*S + s
    int s = row & (S_LEN - 1);
    size_t base = (size_t)row * HEADD;
    float x1 = (float)t[base + d];
    float x2 = (float)t[base + d + 64];
    float c1 = cosb[s * HEADD + d],      s1 = sinb[s * HEADD + d];
    float c2 = cosb[s * HEADD + d + 64], s2 = sinb[s * HEADD + d + 64];
    t[base + d]      = (bf16_t)(x1 * c1 - x2 * s1);
    t[base + d + 64] = (bf16_t)(x2 * c2 + x1 * s2);
}

// ---------------------------------------------------------------------------
// Flash-style attention: one block = one (b,h) and 64 query rows.
// Q,K: [b,h,s,d] bf16; Vt: [b,h,d,s] bf16; ctx out: [b,s,h*d] bf16.
// K tiles double-buffered via async global->LDS; peeled tail.
// ---------------------------------------------------------------------------
__global__ void __launch_bounds__(256) flash_attn_kernel(const bf16_t* __restrict__ Q,
                                                         const bf16_t* __restrict__ Kk,
                                                         const bf16_t* __restrict__ Vt,
                                                         const float* __restrict__ mask,
                                                         bf16_t* __restrict__ ctx) {
    __shared__ __align__(16) bf16_t Qs[64 * 128];
    __shared__ __align__(16) bf16_t Ks[2][64 * 128];
    __shared__ __align__(16) float  Sf[64 * 64];
    __shared__ __align__(16) bf16_t Ps[64 * 64];
    __shared__ float m_s[64], l_s[64], a_s[64];

    const int tid  = threadIdx.x;
    const int lane = tid & 31;
    const int wave = tid >> 5;
    const int q0   = blockIdx.x * 64;
    const int bh   = blockIdx.y;             // b*NH + h
    const int b    = bh / NHEADS;
    const int h    = bh % NHEADS;

    const bf16_t* Qbase = Q  + (size_t)bh * S_LEN * HEADD;
    const bf16_t* Kbase = Kk + (size_t)bh * S_LEN * HEADD;
    const bf16_t* Vbase = Vt + (size_t)bh * HEADD * S_LEN;
    const float*  mbase = mask + (size_t)b * S_LEN * S_LEN;

    // staging assignment: 64x128 tile, 4 x b128 per thread
    const int strow = tid >> 2;
    const int stcol = (tid & 3) * 32;

    auto issue_ktile = [&](int kt, int buf) {
        const bf16_t* ksrc = &Kbase[(size_t)(kt + strow) * HEADD + stcol];
        bf16_t* kdst = &Ks[buf][strow * 128 + stcol];
        async_ld_b128(kdst,      ksrc);
        async_ld_b128(kdst + 8,  ksrc + 8);
        async_ld_b128(kdst + 16, ksrc + 16);
        async_ld_b128(kdst + 24, ksrc + 24);
    };

    // async-stage Q tile and first K tile
    {
        const bf16_t* qsrc = &Qbase[(size_t)(q0 + strow) * HEADD + stcol];
        bf16_t* qdst = &Qs[strow * 128 + stcol];
        async_ld_b128(qdst,      qsrc);
        async_ld_b128(qdst + 8,  qsrc + 8);
        async_ld_b128(qdst + 16, qsrc + 16);
        async_ld_b128(qdst + 24, qsrc + 24);
        issue_ktile(0, 0);
    }
    if (tid < 64) { m_s[tid] = -__builtin_inff(); l_s[tid] = 0.f; }

    // output accumulators: wave -> rows tile mt_o, d tiles dtb..dtb+3
    const int mt_o = wave & 3;
    const int dtb  = (wave >> 2) * 4;
    v8f oacc[4] = {{}, {}, {}, {}};

    // score tile assignment
    const int mt  = wave & 3;
    const int ntb = (wave >> 2) * 2;
    const float scale = 0.08838834764831845f;   // 1/sqrt(128)

    auto process_tile = [&](int kt, int cur, bool has_next) {
        // scores: S = Q(64x128) @ K^T(128x64), 4 k-steps of 32
        v8f sc0 = {}, sc1 = {};
#pragma unroll
        for (int kb = 0; kb < 128; kb += 32) {
            int am = mt * 16 + (lane & 15);
            int ak = kb + (lane >> 4) * 16;
            v16bf a = bf_cat(*(v8bf*)&Qs[am * 128 + ak],
                             *(v8bf*)&Qs[am * 128 + ak + 8]);
            int bk  = kb + (lane >> 4) * 16;
            int bn0 = (ntb + 0) * 16 + (lane & 15);
            int bn1 = (ntb + 1) * 16 + (lane & 15);
            v16bf b0 = bf_cat(*(v8bf*)&Ks[cur][bn0 * 128 + bk], *(v8bf*)&Ks[cur][bn0 * 128 + bk + 8]);
            v16bf b1 = bf_cat(*(v8bf*)&Ks[cur][bn1 * 128 + bk], *(v8bf*)&Ks[cur][bn1 * 128 + bk + 8]);
            sc0 = wmma_bf16(a, b0, sc0);
            sc1 = wmma_bf16(a, b1, sc1);
        }
        {
            int rbase = mt * 16 + (lane >> 4) * 8;
            int c0 = (ntb + 0) * 16 + (lane & 15);
            int c1 = (ntb + 1) * 16 + (lane & 15);
#pragma unroll
            for (int r = 0; r < 8; ++r) {
                Sf[(rbase + r) * 64 + c0] = sc0[r] * scale;
                Sf[(rbase + r) * 64 + c1] = sc1[r] * scale;
            }
        }
        __syncthreads();

        // online softmax: 4 lanes per row, 16 cols each
        {
            int row = tid >> 2;
            int cq  = (tid & 3) * 16;
            const float* mrow = &mbase[(size_t)(q0 + row) * S_LEN + kt + cq];
            float vloc[16];
            float vmax = -__builtin_inff();
#pragma unroll
            for (int j = 0; j < 16; ++j) {
                float v = Sf[row * 64 + cq + j] + mrow[j];
                vloc[j] = v;
                vmax = fmaxf(vmax, v);
            }
            vmax = fmaxf(vmax, __shfl_xor(vmax, 1, 4));
            vmax = fmaxf(vmax, __shfl_xor(vmax, 2, 4));
            float mprev = m_s[row];
            float mnew  = fmaxf(mprev, vmax);
            float sum = 0.f;
#pragma unroll
            for (int j = 0; j < 16; ++j) {
                float p = __expf(vloc[j] - mnew);
                sum += p;
                Ps[row * 64 + cq + j] = (bf16_t)p;
            }
            sum += __shfl_xor(sum, 1, 4);
            sum += __shfl_xor(sum, 2, 4);
            if ((tid & 3) == 0) {
                float alpha = __expf(mprev - mnew);
                a_s[row] = alpha;
                l_s[row] = l_s[row] * alpha + sum;
                m_s[row] = mnew;
            }
        }
        __syncthreads();

        // rescale O and accumulate P(64x64) @ Vt-tile(64x128)
        {
            int rbase = mt_o * 16 + (lane >> 4) * 8;
#pragma unroll
            for (int r = 0; r < 8; ++r) {
                float al = a_s[rbase + r];
                oacc[0][r] *= al; oacc[1][r] *= al; oacc[2][r] *= al; oacc[3][r] *= al;
            }
            if (has_next)
                __builtin_prefetch(&Vbase[(size_t)(dtb * 16 + (lane & 15)) * S_LEN + kt + 64], 0, 0);
#pragma unroll
            for (int kb = 0; kb < 64; kb += 32) {
                int am = mt_o * 16 + (lane & 15);
                int ak = kb + (lane >> 4) * 16;
                v16bf a = bf_cat(*(v8bf*)&Ps[am * 64 + ak],
                                 *(v8bf*)&Ps[am * 64 + ak + 8]);
                int kk = kt + kb + (lane >> 4) * 16;
#pragma unroll
                for (int j = 0; j < 4; ++j) {
                    int d = (dtb + j) * 16 + (lane & 15);
                    const v8bf* vp = (const v8bf*)&Vbase[(size_t)d * S_LEN + kk];
                    v16bf bvv = bf_cat(vp[0], vp[1]);
                    oacc[j] = wmma_bf16(a, bvv, oacc[j]);
                }
            }
        }
        __syncthreads();
    };

    int kt = 0;
    for (; kt < S_LEN - 64; kt += 64) {
        const int cur = (kt >> 6) & 1;
        issue_ktile(kt + 64, cur ^ 1);
        wait_async4();
        __syncthreads();
        process_tile(kt, cur, true);
    }
    // peeled tail
    {
        const int cur = (kt >> 6) & 1;
        wait_async0();
        __syncthreads();
        process_tile(kt, cur, false);
    }

    // epilogue: normalize by l and write ctx [b, s, h*128 + d] bf16
    {
        int rbase = mt_o * 16 + (lane >> 4) * 8;
#pragma unroll
        for (int r = 0; r < 8; ++r) {
            int row  = rbase + r;
            float inv = 1.f / l_s[row];
            size_t obase = ((size_t)b * S_LEN + q0 + row) * HDIM + h * HEADD;
#pragma unroll
            for (int j = 0; j < 4; ++j) {
                int d = (dtb + j) * 16 + (lane & 15);
                ctx[obase + d] = (bf16_t)(oacc[j][r] * inv);
            }
        }
    }
}

// ---------------------------------------------------------------------------
// Launch
// ---------------------------------------------------------------------------
extern "C" void kernel_launch(void* const* d_in, const int* in_sizes, int n_in,
                              void* d_out, int out_size, void* d_ws, size_t ws_size,
                              hipStream_t stream) {
    const float* hidden = (const float*)d_in[0];
    const float* cosb   = (const float*)d_in[1];
    const float* sinb   = (const float*)d_in[2];
    const float* mask   = (const float*)d_in[3];
    const float* Wq     = (const float*)d_in[4];
    const float* Wk     = (const float*)d_in[5];
    const float* Wv     = (const float*)d_in[6];
    const float* Wo     = (const float*)d_in[7];
    float* out = (float*)d_out;

    const int M = BATCH * S_LEN;           // 4096
    const int N = HDIM;                    // 2048
    const int K = HDIM;                    // 2048
    const size_t hidN = (size_t)M * HDIM;  // 8388608
    const size_t wN   = (size_t)HDIM * HDIM;
    const size_t headN = (size_t)BATCH * NHEADS * S_LEN * HEADD;  // 8388608

    char* ws = (char*)d_ws;
    size_t off = 0;
    bf16_t* hiddenB = (bf16_t*)(ws + off); off += hidN * sizeof(bf16_t);
    bf16_t* WqB     = (bf16_t*)(ws + off); off += wN * sizeof(bf16_t);
    bf16_t* WkB     = (bf16_t*)(ws + off); off += wN * sizeof(bf16_t);
    bf16_t* WvB     = (bf16_t*)(ws + off); off += wN * sizeof(bf16_t);
    bf16_t* WoB     = (bf16_t*)(ws + off); off += wN * sizeof(bf16_t);
    bf16_t* Qb      = (bf16_t*)(ws + off); off += headN * sizeof(bf16_t);
    bf16_t* Kb      = (bf16_t*)(ws + off); off += headN * sizeof(bf16_t);
    bf16_t* Vtb     = (bf16_t*)(ws + off); off += headN * sizeof(bf16_t);
    bf16_t* ctxB    = (bf16_t*)(ws + off); off += hidN * sizeof(bf16_t);

    // 1) convert fp32 -> bf16
    f32_to_bf16_kernel<<<(int)(hidN / (256 * 4)), 256, 0, stream>>>(hidden, hiddenB, (int)hidN);
    f32_to_bf16_kernel<<<(int)(wN / (256 * 4)), 256, 0, stream>>>(Wq, WqB, (int)wN);
    f32_to_bf16_kernel<<<(int)(wN / (256 * 4)), 256, 0, stream>>>(Wk, WkB, (int)wN);
    f32_to_bf16_kernel<<<(int)(wN / (256 * 4)), 256, 0, stream>>>(Wv, WvB, (int)wN);
    f32_to_bf16_kernel<<<(int)(wN / (256 * 4)), 256, 0, stream>>>(Wo, WoB, (int)wN);

    // 2) QKV projections (128x64 tiles)
    dim3 ggrid(N / 64, M / 128);
    gemm_wmma_kernel<0><<<ggrid, 256, 0, stream>>>(hiddenB, WqB, (void*)Qb,  M, N, K);
    gemm_wmma_kernel<0><<<ggrid, 256, 0, stream>>>(hiddenB, WkB, (void*)Kb,  M, N, K);
    gemm_wmma_kernel<2><<<ggrid, 256, 0, stream>>>(hiddenB, WvB, (void*)Vtb, M, N, K);

    // 3) RoPE on Q and K
    int rope_total = BATCH * NHEADS * S_LEN * 64;
    rope_kernel<<<rope_total / 256, 256, 0, stream>>>(Qb, cosb, sinb, rope_total);
    rope_kernel<<<rope_total / 256, 256, 0, stream>>>(Kb, cosb, sinb, rope_total);

    // 4) attention
    dim3 agrid(S_LEN / 64, BATCH * NHEADS);
    flash_attn_kernel<<<agrid, 256, 0, stream>>>(Qb, Kb, Vtb, mask, ctxB);

    // 5) output projection (f32 out)
    gemm_wmma_kernel<1><<<ggrid, 256, 0, stream>>>(ctxB, WoB, (void*)out, M, N, K);
}
